// AttentionModule_59949153517631
// MI455X (gfx1250) — matrix operand order repaired and checked by reading, not compiled, not run on hardware
//
#include <hip/hip_runtime.h>
#include <hip/hip_bf16.h>
#include <math.h>

// MI455X / gfx1250, wave32. bf16 WMMA path: V_WMMA_F32_16X16X32_BF16.

typedef __attribute__((ext_vector_type(16))) __bf16 v16bf;
typedef __attribute__((ext_vector_type(8)))  __bf16 v8bf;
typedef __attribute__((ext_vector_type(8)))  float  v8f;

#define D_DIM 512
#define S_DIM 2048
#define B_DIM 64
#define KSTEPS (D_DIM / 32)   // 16 k-steps of 32
#define NTILES (D_DIM / 16)   // 32 n-tiles of 16

static __device__ __forceinline__ unsigned pack_bf16(float a, float b) {
  __bf16 x = (__bf16)a;
  __bf16 y = (__bf16)b;
  unsigned short ux = __builtin_bit_cast(unsigned short, x);
  unsigned short uy = __builtin_bit_cast(unsigned short, y);
  return (unsigned)ux | ((unsigned)uy << 16);
}

// ---------------------------------------------------------------------------
// Kernel 0: pack Wr_w [D,D] f32 -> bf16 B-fragments in exact per-lane order.
// Fragment (ks, nt): lane holds column n = nt*16 + (lane&15); lanes 0-15 carry
// K = ks*32+0..15, lanes 16-31 carry K = ks*32+16..31 (16 bf16 per lane).
// ---------------------------------------------------------------------------
__global__ void pack_wr_bf16(const float* __restrict__ Wr, unsigned* __restrict__ wb) {
  int id = blockIdx.x * blockDim.x + threadIdx.x;   // 16384 work-items
  if (id >= KSTEPS * NTILES * 32) return;
  int lane = id & 31;
  int nt   = (id >> 5) & 31;
  int ks   = id >> 10;
  int n    = nt * 16 + (lane & 15);
  int k0   = ks * 32 + ((lane & 16) ? 16 : 0);
  const float* src = Wr + (size_t)n * D_DIM + k0;
  unsigned* dst = wb + ((size_t)(ks * NTILES + nt) * 32 + lane) * 8;
#pragma unroll
  for (int j = 0; j < 8; ++j)
    dst[j] = pack_bf16(src[2 * j], src[2 * j + 1]);
}

// ---------------------------------------------------------------------------
// Kernel 1: q = query @ Wq_w.T + Wq_b   (64x512x512, trivial)
// ---------------------------------------------------------------------------
__global__ void q_gemm(const float* __restrict__ query, const float* __restrict__ Wq,
                       const float* __restrict__ bq, float* __restrict__ qout) {
  int id = blockIdx.x * blockDim.x + threadIdx.x;   // 32768
  int b = id >> 9;
  int e = id & 511;
  const float4* qa = (const float4*)(query + (size_t)b * D_DIM);
  const float4* wa = (const float4*)(Wq + (size_t)e * D_DIM);
  float s = 0.f;
#pragma unroll 4
  for (int i = 0; i < D_DIM / 4; ++i) {
    float4 x = qa[i], w = wa[i];
    s += x.x * w.x + x.y * w.y + x.z * w.z + x.w * w.w;
  }
  qout[id] = s + bq[e];
}

// ---------------------------------------------------------------------------
// Kernel 2: fused r-GEMM + transpose-store + tanh-logit reduction.
// One WG = 16 rows of (B*S); 8 waves x 4 n-tiles cover N = 512.
// ---------------------------------------------------------------------------
__launch_bounds__(256)
__global__ void attn_main(const float* __restrict__ ref,
                          const unsigned* __restrict__ wb,
                          const float* __restrict__ qbuf,
                          const float* __restrict__ Wr_b,
                          const float* __restrict__ value,
                          float* __restrict__ refOut,
                          float* __restrict__ logits) {
  // 16 rows x 512 K as bf16, row stride 260 dwords (520 bf16) -> conflict-free
  __shared__ unsigned ldsA[16 * 260];
  __shared__ float accL[16];

  const int t    = threadIdx.x;
  const int lane = t & 31;
  const int wave = t >> 5;
  const int m0   = blockIdx.x * 16;
  const int b    = m0 / S_DIM;
  const int s0   = m0 % S_DIM;

  // Stage A-panel: 256 threads, each converts 32 consecutive f32 -> 16 dwords
  {
    int row = t >> 4;             // 0..15
    int c0  = (t & 15) * 32;      // 0..480
    const float* src = ref + (size_t)(m0 + row) * D_DIM + c0;
    unsigned* dst = ldsA + row * 260 + (c0 >> 1);
#pragma unroll
    for (int j = 0; j < 16; ++j)
      dst[j] = pack_bf16(src[2 * j], src[2 * j + 1]);
  }
  if (t < 16) accL[t] = 0.f;
  __syncthreads();

  // A fragment base for this lane (contiguous K order, same as the B pack):
  // lanes 0-15: M=lane, K runs {0..15}; lanes 16-31: M=lane-16, K {16..31}.
  const char* aRowBase =
      (const char*)ldsA + (size_t)(lane & 15) * 1040 + ((lane & 16) ? 32 : 0);

  union AU { v16bf v; v8bf h[2]; };

  v8f acc[4];
  const v8f zero = {0.f, 0.f, 0.f, 0.f, 0.f, 0.f, 0.f, 0.f};
  acc[0] = zero; acc[1] = zero; acc[2] = zero; acc[3] = zero;

  const int ntb = wave * 4;   // this wave's first n-tile

  for (int ks = 0; ks < KSTEPS; ++ks) {
    AU a;
    a.h[0] = *(const v8bf*)(aRowBase + ks * 64);        // K = ks*32 + off + 0..7
    a.h[1] = *(const v8bf*)(aRowBase + ks * 64 + 16);   // K = ks*32 + off + 8..15
#pragma unroll
    for (int nt = 0; nt < 4; ++nt) {
      const v16bf bfrag = *(const v16bf*)(
          wb + ((size_t)(ks * NTILES + ntb + nt) * 32 + lane) * 8);
      acc[nt] = __builtin_amdgcn_wmma_f32_16x16x32_bf16(
          false, a.v, false, bfrag, (short)0, acc[nt], false, false);
    }
  }

  // Epilogue: C/D layout -> lane gives N=lane&15 (+tile), VGPR g gives
  // s_local = g + 8*(lane>>4). Per-lane 8 consecutive S positions.
  float p[8] = {0.f, 0.f, 0.f, 0.f, 0.f, 0.f, 0.f, 0.f};
  const int nlo  = lane & 15;
  const int soff = (lane >> 4) * 8;

#pragma unroll
  for (int nt = 0; nt < 4; ++nt) {
    const int n = (ntb + nt) * 16 + nlo;
    const float rb = Wr_b[n];
    const float qv = qbuf[(size_t)b * D_DIM + n];
    const float vv = value[n];
    float r[8];
#pragma unroll
    for (int g = 0; g < 8; ++g) {
      r[g] = acc[nt][g] + rb;
      p[g] += tanhf(qv + r[g]) * vv;
    }
    float* outp = refOut + ((size_t)b * D_DIM + n) * S_DIM + (s0 + soff);
    float4 lo = {r[0], r[1], r[2], r[3]};
    float4 hi = {r[4], r[5], r[6], r[7]};
    *(float4*)(outp)     = lo;
    *(float4*)(outp + 4) = hi;
  }

  // Reduce over N: tree across the 16 lanes of each half (halves own
  // disjoint S ranges, xor<16 keeps lanes within their half).
#pragma unroll
  for (int m = 1; m < 16; m <<= 1) {
#pragma unroll
    for (int g = 0; g < 8; ++g)
      p[g] += __shfl_xor(p[g], m, 32);
  }
  if (nlo == 0) {
#pragma unroll
    for (int g = 0; g < 8; ++g)
      atomicAdd(&accL[soff + g], p[g]);   // ds_add_f32 across waves
  }
  __syncthreads();

  if (t < 16)
    logits[(size_t)b * S_DIM + (s0 + t)] = 10.0f * tanhf(accL[t]);
}

// ---------------------------------------------------------------------------
extern "C" void kernel_launch(void* const* d_in, const int* in_sizes, int n_in,
                              void* d_out, int out_size, void* d_ws, size_t ws_size,
                              hipStream_t stream) {
  (void)in_sizes; (void)n_in; (void)out_size; (void)ws_size;
  const float* query = (const float*)d_in[0];
  const float* ref   = (const float*)d_in[1];
  const float* Wq_w  = (const float*)d_in[2];
  const float* Wq_b  = (const float*)d_in[3];
  const float* Wr_w  = (const float*)d_in[4];
  const float* Wr_b  = (const float*)d_in[5];
  const float* value = (const float*)d_in[6];

  float* out    = (float*)d_out;
  float* refOut = out;                                       // [B, D, S]
  float* logits = out + (size_t)B_DIM * D_DIM * S_DIM;       // [B, S]

  unsigned* wb = (unsigned*)d_ws;                            // 512 KB bf16 W pack
  float* qbuf  = (float*)((char*)d_ws + 524288);             // 128 KB q buffer

  pack_wr_bf16<<<64, 256, 0, stream>>>(Wr_w, wb);
  q_gemm<<<128, 256, 0, stream>>>(query, Wq_w, Wq_b, qbuf);
  attn_main<<<(B_DIM * S_DIM) / 16, 256, 0, stream>>>(ref, wb, qbuf, Wr_b, value,
                                                      refOut, logits);
}
